// SVM_B1_9062380994856
// MI455X (gfx1250) — compile-verified, matrix operand order checked by reading, not run on hardware
//
#include <hip/hip_runtime.h>
#include <hip/hip_bf16.h>

typedef __attribute__((ext_vector_type(16))) __bf16 v16bf;
typedef __attribute__((ext_vector_type(8)))  __bf16 v8bf;
typedef __attribute__((ext_vector_type(4)))  __bf16 v4bf;
typedef __attribute__((ext_vector_type(8)))  float  v8f;

#define K_TOTAL 65536
#define N_PAD   512
#define B_ROWS  64
#define NTILES  32
#define KCHUNKS 16
#define KCHUNK  (K_TOTAL / KCHUNKS)   // 4096
#define KSLAB   256
#define NSLABS  (KCHUNK / KSLAB)      // 16
#define KSTEPS  (KSLAB / 32)          // 8
#define LDA     (KSLAB + 8)           // 264 elems = 528B row stride -> conflict-free b128 reads
#define GAMMA_F 1e-4f

__device__ __forceinline__ __bf16 f2bf(float f) { return (__bf16)f; }

__global__ void __launch_bounds__(128)
svm_zero_ws(float* ws, int n) {
    int i = blockIdx.x * blockDim.x + threadIdx.x;
    if (i < n) ws[i] = 0.0f;
}

// grid = 32 ntiles * 16 kchunks = 512 blocks, 128 threads (4 waves)
__global__ void __launch_bounds__(128)
svm_gemm(const float* __restrict__ x, const float* __restrict__ centers,
         float* __restrict__ ws) {
    __shared__ __align__(16) __bf16 sA[B_ROWS * LDA];  // ~33 KB
    __shared__ __align__(16) __bf16 sB[16 * LDA];      // ~8.3 KB

    float* cross = ws;                         // [64][512] partial GEMM accumulator
    float* c_sq  = ws + B_ROWS * N_PAD;        // [512]
    float* x_sq  = c_sq + N_PAD;               // [64]

    const int tid   = threadIdx.x;
    const int lane  = tid & 31;
    const int wave  = tid >> 5;
    const int ntile = blockIdx.x & (NTILES - 1);
    const int kc    = blockIdx.x >> 5;
    const long k0   = (long)kc * KCHUNK;

    // --- B staging role: thread -> (center row, float4 column) ---
    const int  bn     = tid >> 3;                  // 0..15 local center
    const int  bq     = tid & 7;
    const int  n_glob = ntile * 16 + bn;
    const bool n_ok   = (n_glob < 500);
    const float4* Bsrc =
        (const float4*)(centers + (long)(n_ok ? n_glob : 0) * K_TOTAL + k0);

    // --- A staging role ---
    const int ar = tid >> 1;                       // 0..63 batch row
    const int ah = tid & 1;
    const float4* Asrc = (const float4*)(x + (long)ar * K_TOTAL + k0);

    // --- compute role (ISA 16-bit A/B lane layout) ---
    const int kbase = (lane >> 4) << 3;            // 0 or 8
    const int mrow  = wave * 16 + (lane & 15);
    const int ncol  = lane & 15;

    v8f   acc = {};
    float csq = 0.0f, xsq = 0.0f;

    for (int slab = 0; slab < NSLABS; ++slab) {
        // stage B tile [16 x 256] f32 -> bf16 LDS, fuse ||c||^2 partial
#pragma unroll
        for (int i = 0; i < 8; ++i) {
            const int k4 = bq + 8 * i;
            float4 v = make_float4(0.f, 0.f, 0.f, 0.f);
            if (n_ok) v = Bsrc[slab * (KSLAB / 4) + k4];
            csq += v.x * v.x + v.y * v.y + v.z * v.z + v.w * v.w;
            v4bf p; p[0] = f2bf(v.x); p[1] = f2bf(v.y); p[2] = f2bf(v.z); p[3] = f2bf(v.w);
            *(v4bf*)&sB[bn * LDA + k4 * 4] = p;
        }
        // stage A tile [64 x 256] f32 -> bf16 LDS, fuse ||x||^2 partial (once per K-chunk)
#pragma unroll
        for (int i = 0; i < 32; ++i) {
            const int k4 = ah + 2 * i;
            float4 v = Asrc[slab * (KSLAB / 4) + k4];
            if (ntile == 0)
                xsq += v.x * v.x + v.y * v.y + v.z * v.z + v.w * v.w;
            v4bf p; p[0] = f2bf(v.x); p[1] = f2bf(v.y); p[2] = f2bf(v.z); p[3] = f2bf(v.w);
            *(v4bf*)&sA[ar * LDA + k4 * 4] = p;
        }
        __syncthreads();

#pragma unroll
        for (int s = 0; s < KSTEPS; ++s) {
            const int ka = s * 32 + kbase;
            v8bf alo = *(const v8bf*)&sA[mrow * LDA + ka];
            v8bf ahi = *(const v8bf*)&sA[mrow * LDA + ka + 16];
            v8bf blo = *(const v8bf*)&sB[ncol * LDA + ka];
            v8bf bhi = *(const v8bf*)&sB[ncol * LDA + ka + 16];
            v16bf a = __builtin_shufflevector(alo, ahi, 0,1,2,3,4,5,6,7,8,9,10,11,12,13,14,15);
            v16bf b = __builtin_shufflevector(blo, bhi, 0,1,2,3,4,5,6,7,8,9,10,11,12,13,14,15);
            acc = __builtin_amdgcn_wmma_f32_16x16x32_bf16(
                      false, a, false, b, (short)0, acc, false, false);
        }
        __syncthreads();
    }

    // C/D layout: VGPR v -> M = v + 8*(lane>=16), N = lane&15
    const int m_base = wave * 16 + (lane >> 4) * 8;
    const int n_out  = ntile * 16 + (lane & 15);
#pragma unroll
    for (int v = 0; v < 8; ++v)
        atomicAdd(&cross[(m_base + v) * N_PAD + n_out], acc[v]);

    if (n_ok)       atomicAdd(&c_sq[n_glob], csq);
    if (ntile == 0) atomicAdd(&x_sq[ar], xsq);
}

// dist^2 = ||x||^2 - 2*cross + ||c||^2 ; rbf = exp(-gamma*dist^2); out = rbf @ fc_w.T + fc_b
__global__ void __launch_bounds__(512)
svm_epilogue(const float* __restrict__ ws, const float* __restrict__ fc_w,
             const float* __restrict__ fc_b, float* __restrict__ out) {
    const int t = blockIdx.x * blockDim.x + threadIdx.x;
    if (t >= B_ROWS * 8) return;
    const int b = t >> 3;
    const int j = t & 7;
    const float* cross = ws;
    const float* c_sq  = ws + B_ROWS * N_PAD;
    const float* x_sq  = c_sq + N_PAD;
    const float  xs    = x_sq[b];
    float accum = fc_b[j];
    for (int n = 0; n < 500; ++n) {
        const float d   = xs - 2.0f * cross[b * N_PAD + n] + c_sq[n];
        accum += __expf(-GAMMA_F * d) * fc_w[j * 500 + n];
    }
    out[t] = accum;
}

extern "C" void kernel_launch(void* const* d_in, const int* in_sizes, int n_in,
                              void* d_out, int out_size, void* d_ws, size_t ws_size,
                              hipStream_t stream) {
    const float* x       = (const float*)d_in[0];   // [64,256,256]
    const float* centers = (const float*)d_in[1];   // [500,65536]
    const float* fc_w    = (const float*)d_in[2];   // [8,500]
    const float* fc_b    = (const float*)d_in[3];   // [8]
    float* out = (float*)d_out;                     // [64,8]
    float* ws  = (float*)d_ws;

    const int ws_elems = B_ROWS * N_PAD + N_PAD + B_ROWS;  // cross + c_sq + x_sq
    svm_zero_ws<<<(ws_elems + 127) / 128, 128, 0, stream>>>(ws, ws_elems);
    svm_gemm<<<NTILES * KCHUNKS, 128, 0, stream>>>(x, centers, ws);
    svm_epilogue<<<1, 512, 0, stream>>>(ws, fc_w, fc_b, out);
}
